// RepLKNetTSMV2R_9294309229049
// MI455X (gfx1250) — compile-verified, matrix operand order checked by reading, not run on hardware
//
#include <hip/hip_runtime.h>

typedef unsigned short u16;
typedef unsigned int   u32;
typedef __attribute__((ext_vector_type(16))) __bf16 v16bf;
typedef __attribute__((ext_vector_type(8)))  float  v8f;
typedef __attribute__((ext_vector_type(4)))  int    v4i;

#define HW    3136
#define WD    56
#define CC    256
#define FFND  1024
#define NIMG  16
#define TSEG  8

#if __has_builtin(__builtin_amdgcn_global_load_async_to_lds_b128) && \
    __has_builtin(__builtin_amdgcn_s_wait_asynccnt)
#define USE_ASYNC_LDS 1
#else
#define USE_ASYNC_LDS 0
#endif

typedef __attribute__((address_space(1))) v4i* g4_ptr;   // global <4 x i32>*
typedef __attribute__((address_space(3))) v4i* l4_ptr;   // LDS <4 x i32>*

__device__ __forceinline__ g4_ptr to_glb4(const void* p) {
    return (g4_ptr)(unsigned long long)p;
}
__device__ __forceinline__ l4_ptr to_lds4(const void* p) {
    return (l4_ptr)(u32)(unsigned long long)p;   // generic LDS addr low 32b = LDS offset
}

__device__ __forceinline__ u16 f2bf(float f) {
    u32 u = __float_as_uint(f);
    u32 r = u + 0x7FFFu + ((u >> 16) & 1u);   // round-to-nearest-even
    return (u16)(r >> 16);
}

// ---------------------------------------------------------------------------
// BN folding: s = g*rsqrt(v+eps), t = b - m*s, for all 8 BN packs.
// bnbuf float layout:
//   0: s_pre0  256: t_pre0   512: s_pw1  768: t_pw1
//   1024: s_lk 1280: t_lk    1536: s_sk  1792: t_sk
//   2048: s_pw2 2304: t_pw2  2560: s_preffn 2816: t_preffn
//   3072: s_f1(1024) 4096: t_f1(1024)  5120: s_f2 5376: t_f2
// ---------------------------------------------------------------------------
__global__ void bnprep_kernel(const float* p_pre, const float* p_pw1, const float* p_lk,
                              const float* p_sk, const float* p_pw2, const float* p_preffn,
                              const float* p_f1, const float* p_f2, float* bnbuf) {
    int c = threadIdx.x;
    auto doPack = [&](const float* p, int C, int off) {
        float g = p[c], b = p[C + c], m = p[2 * C + c], v = p[3 * C + c];
        float s = g * rsqrtf(v + 1e-5f);
        bnbuf[off + c]     = s;
        bnbuf[off + C + c] = b - m * s;
    };
    if (c < 256) {
        doPack(p_pre,    256, 0);
        doPack(p_pw1,    256, 512);
        doPack(p_lk,     256, 1024);
        doPack(p_sk,     256, 1536);
        doPack(p_pw2,    256, 2048);
        doPack(p_preffn, 256, 2560);
        doPack(p_f2,     256, 5120);
    }
    if (c < 1024) doPack(p_f1, 1024, 3072);
}

// ---------------------------------------------------------------------------
// Convert the four pointwise weight matrices to bf16 ([M,K] row-major kept).
// ---------------------------------------------------------------------------
__global__ void wconv_kernel(const float* w_pw1, const float* w_pw2, const float* w_f1,
                             const float* w_f2, u16* o_pw1, u16* o_pw2, u16* o_f1, u16* o_f2) {
    int i = blockIdx.x * blockDim.x + threadIdx.x;
    if      (i < 65536)  o_pw1[i]          = f2bf(w_pw1[i]);
    else if (i < 131072) o_pw2[i - 65536]  = f2bf(w_pw2[i - 65536]);
    else if (i < 393216) o_f1[i - 131072]  = f2bf(w_f1[i - 131072]);
    else if (i < 655360) o_f2[i - 393216]  = f2bf(w_f2[i - 393216]);
}

// ---------------------------------------------------------------------------
// TSM temporal shift + prelkb BN -> bf16, vectorized over 4 pixels.
// ---------------------------------------------------------------------------
__global__ void shift_bn_kernel(const float* __restrict__ x, const float* __restrict__ bnbuf,
                                u16* __restrict__ y0) {
    int i = blockIdx.x * blockDim.x + threadIdx.x;
    const int TOT4 = NIMG * CC * (HW / 4);
    if (i >= TOT4) return;
    int hw4 = i % (HW / 4);
    int c   = (i / (HW / 4)) % CC;
    int n   = i / ((HW / 4) * CC);
    int t   = n & (TSEG - 1);
    const int fold = CC / 8;
    float4 v = make_float4(0.f, 0.f, 0.f, 0.f);
    int  nsrc  = n;
    bool valid = true;
    if      (c < fold)     { nsrc = n + 1; valid = (t != TSEG - 1); }  // shift backward
    else if (c < 2 * fold) { nsrc = n - 1; valid = (t != 0); }         // shift forward
    if (valid)
        v = ((const float4*)x)[((size_t)nsrc * CC + c) * (HW / 4) + hw4];
    float s = bnbuf[c], tt = bnbuf[256 + c];
    u32 lo = (u32)f2bf(v.x * s + tt) | ((u32)f2bf(v.y * s + tt) << 16);
    u32 hi = (u32)f2bf(v.z * s + tt) | ((u32)f2bf(v.w * s + tt) << 16);
    ((uint2*)y0)[i] = make_uint2(lo, hi);
}

// ---------------------------------------------------------------------------
// Fused depthwise 31x31 + 5x5 + BN(lk)+BN(sk) + ReLU.  One block per
// (n,c,h-tile of 14).  Input patch (44x86) and per-channel weights in LDS.
// ---------------------------------------------------------------------------
__global__ __launch_bounds__(256) void dwconv_kernel(const float* __restrict__ y1,
                                                     const float* __restrict__ lkw,
                                                     const float* __restrict__ skw,
                                                     const float* __restrict__ bnbuf,
                                                     u16* __restrict__ y2) {
    __shared__ float patch[44 * 87];
    __shared__ float wl[961];
    __shared__ float w5[25];

    int nc  = blockIdx.x;
    int c   = nc % CC, n = nc / CC;
    int h0  = blockIdx.y * 14;
    int tid = threadIdx.x;

    const float* src = y1 + ((size_t)n * CC + c) * HW;
    for (int i = tid; i < 44 * 86; i += 256) {
        int py = i / 86, px = i % 86;
        int gy = h0 + py - 15, gx = px - 15;
        float v = 0.f;
        if (gy >= 0 && gy < WD && gx >= 0 && gx < WD) v = src[gy * WD + gx];
        patch[py * 87 + px] = v;
    }
    for (int i = tid; i < 961; i += 256) wl[i] = lkw[c * 961 + i];
    if (tid < 25) w5[tid] = skw[c * 25 + tid];
    __syncthreads();

    float slk = bnbuf[1024 + c], tlk = bnbuf[1280 + c];
    float ssk = bnbuf[1536 + c], tsk = bnbuf[1792 + c];
    u16* dst = y2 + ((size_t)n * CC + c) * HW;

    for (int i = tid; i < 14 * WD; i += 256) {
        int oy = i / WD, ox = i % WD;
        float acc = 0.f;
        for (int ky = 0; ky < 31; ++ky) {
            const float* pr = &patch[(oy + ky) * 87 + ox];
            const float* wr = &wl[ky * 31];
            #pragma unroll
            for (int kx = 0; kx < 31; ++kx) acc = fmaf(wr[kx], pr[kx], acc);
        }
        float a5 = 0.f;
        #pragma unroll
        for (int ky = 0; ky < 5; ++ky) {
            const float* pr = &patch[(oy + 13 + ky) * 87 + ox + 13];
            const float* wr = &w5[ky * 5];
            #pragma unroll
            for (int kx = 0; kx < 5; ++kx) a5 = fmaf(wr[kx], pr[kx], a5);
        }
        float v = slk * acc + tlk + ssk * a5 + tsk;
        dst[(h0 + oy) * WD + ox] = f2bf(fmaxf(v, 0.f));
    }
}

// ---------------------------------------------------------------------------
// WMMA BF16 GEMM:  out[o,p] = epilogue( sum_c W[o,c]*act[c,p] )
// act layout [n][K][3136] (bf16 or f32), W [M,K] bf16 row-major.
// Block: 128(M) x 64(N), K-step 32, double-buffered LDS, 8 waves,
// each wave a 2x2 grid of v_wmma_f32_16x16x32_bf16 tiles.
// Weight tiles are staged with GLOBAL_LOAD_ASYNC_TO_LDS_B128 (ASYNCcnt) when
// the builtin exists; A-tile LDS stride = 20 u32 (80B) so every per-lane
// async destination is 16B-aligned and fragment gathers stay conflict-free.
// EPI: 0 = BN+ReLU->f32 ; 1 = BN+residual->f32 ; 2 = BN+GELU->bf16
// ---------------------------------------------------------------------------
#define ASTRIDE 20
template <int EPI, bool IN_BF16, bool IN_SCALE>
__global__ __launch_bounds__(256) void gemm_wmma_kernel(
    const void* __restrict__ act, const u16* __restrict__ wgt,
    const float* __restrict__ bn_s, const float* __restrict__ bn_t,
    const float* __restrict__ in_s, const float* __restrict__ in_t,
    const float* __restrict__ resid, void* __restrict__ outp,
    int M, int K)
{
    __shared__ __align__(16) u32 ldsA[2][128 * ASTRIDE];  // weights, row-major (M x 32 bf16)
    __shared__ u32 ldsB[2][64 * 17];                      // activations, col-major k-pairs

    const int tid   = threadIdx.x;
    const int bx    = blockIdx.x;
    const int n_img = bx / 49;
    const int hw0   = (bx % 49) * 64;   // 49*64 == 3136
    const int o0    = blockIdx.y * 128;
    const int KT    = K >> 5;

    // staging roles
    const int ar = tid >> 1;   // weight row 0..127
    const int ah = tid & 1;    // which 32B half of the 64B row
    const int bk = tid >> 3;   // activation channel row 0..31
    const int bs = tid & 7;    // 8-pixel column segment

    uint4  ra0, ra1;           // weight regs (fallback path)
    uint4  rb_b;               // bf16 activation regs
    float4 rf0, rf1;           // f32 activation regs
    float  scl = 1.f, sht = 0.f;

    // ---- A tile: async DMA straight into LDS (or reg round-trip fallback)
    auto loadA = [&](int kt, int buf) {
        const u16* g = wgt + (size_t)(o0 + ar) * K + kt * 32 + ah * 16;
        u32* l = &ldsA[buf][ar * ASTRIDE + ah * 8];
#if USE_ASYNC_LDS
        __builtin_amdgcn_global_load_async_to_lds_b128(to_glb4(g),     to_lds4(l),     0, 0);
        __builtin_amdgcn_global_load_async_to_lds_b128(to_glb4(g + 8), to_lds4(l + 4), 0, 0);
#else
        const uint4* gA = (const uint4*)g;
        ra0 = gA[0];
        ra1 = gA[1];
        (void)l;
#endif
    };
    auto flushA = [&](int buf) {
#if !USE_ASYNC_LDS
        u32* pa = &ldsA[buf][ar * ASTRIDE + ah * 8];
        pa[0] = ra0.x; pa[1] = ra0.y; pa[2] = ra0.z; pa[3] = ra0.w;
        pa[4] = ra1.x; pa[5] = ra1.y; pa[6] = ra1.z; pa[7] = ra1.w;
#else
        (void)buf;
#endif
    };
    auto waitA = [&]() {
#if USE_ASYNC_LDS
        __builtin_amdgcn_s_wait_asynccnt(0);
#endif
    };

    // ---- B tile: load + (optional pre-BN) + f32->bf16 + k-pair transpose
    auto loadB = [&](int kt) {
        int    ch = kt * 32 + bk;
        size_t sb = ((size_t)n_img * K + ch) * HW + hw0 + bs * 8;
        if (IN_BF16) {
            rb_b = *(const uint4*)((const u16*)act + sb);
        } else {
            const float4* gB = (const float4*)((const float*)act + sb);
            rf0 = gB[0];
            rf1 = gB[1];
            if (IN_SCALE) { scl = in_s[ch]; sht = in_t[ch]; }
        }
    };
    auto storeB = [&](int buf) {
        u16 h[8];
        if (IN_BF16) {
            const u16* q = (const u16*)&rb_b;
            #pragma unroll
            for (int e = 0; e < 8; ++e) h[e] = q[e];
        } else {
            float f[8] = {rf0.x, rf0.y, rf0.z, rf0.w, rf1.x, rf1.y, rf1.z, rf1.w};
            #pragma unroll
            for (int e = 0; e < 8; ++e) {
                float v = f[e];
                if (IN_SCALE) v = v * scl + sht;
                h[e] = f2bf(v);
            }
        }
        u16* pb = (u16*)&ldsB[buf][0];
        #pragma unroll
        for (int e = 0; e < 8; ++e) {
            int ncol = bs * 8 + e;
            pb[(ncol * 17 + (bk >> 1)) * 2 + (bk & 1)] = h[e];
        }
    };

    const int lane   = tid & 31;
    const int wv     = tid >> 5;
    const int m_base = (wv & 3) * 32;
    const int n_base = (wv >> 2) * 32;
    const int g      = lane >> 4;
    const int ln16   = lane & 15;

    v8f acc[2][2];
    #pragma unroll
    for (int a = 0; a < 2; ++a)
        #pragma unroll
        for (int b = 0; b < 2; ++b)
            #pragma unroll
            for (int e = 0; e < 8; ++e) acc[a][b][e] = 0.f;

    loadA(0, 0);
    loadB(0);
    flushA(0);
    storeB(0);
    waitA();
    __syncthreads();

    for (int kt = 0; kt < KT; ++kt) {
        if (kt + 1 < KT) {
            loadA(kt + 1, (kt + 1) & 1);
            loadB(kt + 1);
        }
        if (kt + 2 < KT)
            __builtin_prefetch((const void*)(wgt + (size_t)(o0 + ar) * K + (kt + 2) * 32), 0, 1);
        const int cur = kt & 1;

        // fragment gathers per ISA 7.12.2 layout:
        //   A: M = lane%16, k-pair(v) = (v&3) + 8*(v>>2) + 4*(lane/16)
        //   B (mirrored): N = lane%16, same k-pair mapping
        union { u32 u[8]; v16bf v; } afr[2], bfr[2];
        #pragma unroll
        for (int t2 = 0; t2 < 2; ++t2) {
            int row = m_base + 16 * t2 + ln16;
            int col = n_base + 16 * t2 + ln16;
            #pragma unroll
            for (int vv = 0; vv < 8; ++vv) {
                int kp = (vv & 3) + ((vv >> 2) << 3) + 4 * g;
                afr[t2].u[vv] = ldsA[cur][row * ASTRIDE + kp];
                bfr[t2].u[vv] = ldsB[cur][col * 17 + kp];
            }
        }
        #pragma unroll
        for (int tm = 0; tm < 2; ++tm)
            #pragma unroll
            for (int tn = 0; tn < 2; ++tn)
                acc[tm][tn] = __builtin_amdgcn_wmma_f32_16x16x32_bf16(
                    false, afr[tm].v, false, bfr[tn].v, (short)0, acc[tm][tn], false, false);

        if (kt + 1 < KT) {
            flushA((kt + 1) & 1);
            storeB((kt + 1) & 1);
        }
        waitA();
        __syncthreads();
    }

    // epilogue: C/D layout  M = i + 8*(lane/16), N = lane%16
    #pragma unroll
    for (int tm = 0; tm < 2; ++tm) {
        #pragma unroll
        for (int tn = 0; tn < 2; ++tn) {
            #pragma unroll
            for (int i = 0; i < 8; ++i) {
                int    o    = o0 + m_base + 16 * tm + i + 8 * g;
                int    p    = hw0 + n_base + 16 * tn + ln16;
                size_t addr = ((size_t)n_img * M + o) * HW + p;
                float  v    = acc[tm][tn][i] * bn_s[o] + bn_t[o];
                if (EPI == 0) {
                    ((float*)outp)[addr] = fmaxf(v, 0.f);
                } else if (EPI == 1) {
                    ((float*)outp)[addr] = v + resid[addr];
                } else {
                    float gl = 0.5f * v * (1.f + erff(v * 0.70710678118f));
                    ((u16*)outp)[addr] = f2bf(gl);
                }
            }
        }
    }
}

// ---------------------------------------------------------------------------
extern "C" void kernel_launch(void* const* d_in, const int* in_sizes, int n_in,
                              void* d_out, int out_size, void* d_ws, size_t ws_size,
                              hipStream_t stream) {
    const float* x       = (const float*)d_in[0];
    const float* prelkb  = (const float*)d_in[1];
    const float* pw1_w   = (const float*)d_in[2];
    const float* pw1_bn  = (const float*)d_in[3];
    const float* lk_w    = (const float*)d_in[4];
    const float* lk_bn   = (const float*)d_in[5];
    const float* sk_w    = (const float*)d_in[6];
    const float* sk_bn   = (const float*)d_in[7];
    const float* pw2_w   = (const float*)d_in[8];
    const float* pw2_bn  = (const float*)d_in[9];
    const float* preffn  = (const float*)d_in[10];
    const float* ffn1_w  = (const float*)d_in[11];
    const float* ffn1_bn = (const float*)d_in[12];
    const float* ffn2_w  = (const float*)d_in[13];
    const float* ffn2_bn = (const float*)d_in[14];

    char*  ws    = (char*)d_ws;
    float* bnbuf = (float*)ws;                                  //   32 KB
    u16*   w1    = (u16*)(ws + 32768);                          //  128 KB
    u16*   w2    = (u16*)(ws + 32768 + 131072);                 //  128 KB
    u16*   wf1   = (u16*)(ws + 32768 + 2 * 131072);             //  512 KB
    u16*   wf2   = (u16*)(ws + 32768 + 2 * 131072 + 524288);    //  512 KB
    char*  ab    = ws + 32768 + 2 * 131072 + 2 * 524288;
    u16*   y0    = (u16*)(ab);                                           // 25.7 MB bf16
    float* y1    = (float*)(ab + 25690112);                              // 51.4 MB f32
    u16*   y2    = (u16*)(ab + 25690112 + 51380224);                     // 25.7 MB bf16
    float* x2    = (float*)(ab + 2 * 25690112ll + 51380224);             // 51.4 MB f32
    u16*   hm    = (u16*)(ab + 2 * 25690112ll + 2 * 51380224ll);         // 102.8 MB bf16

    bnprep_kernel<<<1, 1024, 0, stream>>>(prelkb, pw1_bn, lk_bn, sk_bn, pw2_bn,
                                          preffn, ffn1_bn, ffn2_bn, bnbuf);
    wconv_kernel<<<2560, 256, 0, stream>>>(pw1_w, pw2_w, ffn1_w, ffn2_w, w1, w2, wf1, wf2);
    shift_bn_kernel<<<(3211264 + 255) / 256, 256, 0, stream>>>(x, bnbuf, y0);

    // pw1: relu(bn(W1 @ y0)) -> y1 (f32)
    gemm_wmma_kernel<0, true, false><<<dim3(784, 2), 256, 0, stream>>>(
        y0, w1, bnbuf + 512, bnbuf + 768, nullptr, nullptr, nullptr, y1, 256, 256);

    // fused large-kernel depthwise reparam branch -> y2 (bf16)
    dwconv_kernel<<<dim3(4096, 4), 256, 0, stream>>>(y1, lk_w, sk_w, bnbuf, y2);

    // pw2: bn(W2 @ y2) + x -> x2 (f32)
    gemm_wmma_kernel<1, true, false><<<dim3(784, 2), 256, 0, stream>>>(
        y2, w2, bnbuf + 2048, bnbuf + 2304, nullptr, nullptr, x, x2, 256, 256);

    // ffn1: gelu(bn(Wf1 @ (s_pre*x2+t_pre))) -> hm (bf16)
    gemm_wmma_kernel<2, false, true><<<dim3(784, 8), 256, 0, stream>>>(
        x2, wf1, bnbuf + 3072, bnbuf + 4096, bnbuf + 2560, bnbuf + 2816, nullptr, hm, 1024, 256);

    // ffn2: bn(Wf2 @ hm) + x2 -> d_out (f32)
    gemm_wmma_kernel<1, true, false><<<dim3(784, 2), 256, 0, stream>>>(
        hm, wf2, bnbuf + 5120, bnbuf + 5376, nullptr, nullptr, x2, (float*)d_out, 256, 1024);
}
